// RWKV6TimeMix_38517266710544
// MI455X (gfx1250) — compile-verified, matrix operand order checked by reading, not run on hardware
//
#include <hip/hip_runtime.h>

typedef __bf16 bf16_t;
typedef __attribute__((ext_vector_type(16))) __bf16 v16bf;
typedef __attribute__((ext_vector_type(8)))  __bf16 v8bf;
typedef __attribute__((ext_vector_type(8)))  float  v8f;

static constexpr int Bc = 8, Hc = 16, Tc = 4096, Kc = 64, Vc = 64;
static constexpr int SUB = 16, NSUB = Tc / SUB;
static constexpr long long OUT_ELEMS = (long long)Bc * Hc * Tc * Vc;
#define LOGMIN (-5.2983174f) /* log(0.005) */

// Gather a 16-bit WMMA A/B operand (16x16x32 shape) from a row-major LDS row.
// Per ISA: lane (m,half) holds halves K in [kb+8*half, kb+8*half+8) and
// [kb+16+8*half, kb+16+8*half+8).
__device__ __forceinline__ v16bf load_op(const bf16_t* row, int kb, int half) {
  const v8bf a = *(const v8bf*)(row + kb + 8 * half);
  const v8bf b = *(const v8bf*)(row + kb + 16 + 8 * half);
  return __builtin_shufflevector(a, b, 0, 1, 2, 3, 4, 5, 6, 7,
                                 8, 9, 10, 11, 12, 13, 14, 15);
}

__device__ __forceinline__ v8f wmma_bf16(v16bf a, v16bf b, v8f c) {
  return __builtin_amdgcn_wmma_f32_16x16x32_bf16(false, a, false, b,
                                                 (short)0, c, false, false);
}

__global__ __launch_bounds__(128) void rwkv6_wkv_kernel(
    const float* __restrict__ r, const float* __restrict__ k,
    const float* __restrict__ v, const float* __restrict__ w,
    const float* __restrict__ u, const float* __restrict__ s_in,
    float* __restrict__ out, float* __restrict__ s_out) {
  // Per-tile operand staging (bf16, padded K-dims where the reduction is s=16)
  __shared__ __align__(16) bf16_t rt16[16][64];   // r~[t][k] = r*exp(p)
  __shared__ __align__(16) bf16_t kh16[16][64];   // k^[s][k] = k*exp(-(p+w))
  __shared__ __align__(16) bf16_t ktr[64][32];    // K~ transposed [k][s], s 16..31 = 0
  __shared__ __align__(16) bf16_t vtr[64][32];    // V transposed [v][s], s 16..31 = 0
  __shared__ __align__(16) bf16_t Am[4][16][32];  // masked A per wave [t][s], s 16..31 = 0
  __shared__ __align__(16) bf16_t Sb[4][16][64];  // state mirror bf16, [v][k] per wave
  __shared__ float St[4][16][64];                 // state master f32, [v][k] per wave
  __shared__ float wd[64];                        // exp(sum_t w) per channel
  __shared__ float dpart[4][16];
  __shared__ float diag[16];
  __shared__ float ulds[64];

  const int bh = blockIdx.x;
  const int h = bh & (Hc - 1);
  const int tid = threadIdx.x;
  const int wave = tid >> 5;
  const int lane = tid & 31;
  const int m = lane & 15;
  const int half = lane >> 4;
  const int v0 = wave * 16;

  // ---- init: u, zero pads, state from input ----
  for (int q = tid; q < 64; q += 128) ulds[q] = u[h * Kc + q];
  for (int q = tid; q < 1024; q += 128) {
    int a = q >> 4, s = 16 + (q & 15);
    vtr[a][s] = (bf16_t)0.f;
    ktr[a][s] = (bf16_t)0.f;
  }
  for (int q = tid; q < 1024; q += 128)
    Am[q >> 8][(q >> 4) & 15][16 + (q & 15)] = (bf16_t)0.f;
  for (int q = tid; q < 4096; q += 128) {
    int wv = q >> 10, vl = (q >> 6) & 15, kk = q & 63;
    float val = s_in[(size_t)bh * Kc * Vc + (size_t)kk * Vc + (wv * 16 + vl)];
    St[wv][vl][kk] = val;
    Sb[wv][vl][kk] = (bf16_t)val;
  }
  __syncthreads();

  for (int i = 0; i < NSUB; ++i) {
    const int t0 = i * SUB;

    // ---- phase A: build decayed tiles (threads 0..63: one k-channel each) ----
    if (tid < 64) {
      const int c = tid;
      const size_t base = ((size_t)bh * Tc + t0) * Kc + c;
      float p = 0.f;
      float khf[16];
#pragma unroll
      for (int t = 0; t < 16; ++t) {
        const size_t idx = base + (size_t)t * Kc;
        const float wl = fmaxf(w[idx], LOGMIN);
        const float rv = r[idx], kv = k[idx], vv = v[idx];
        rt16[t][c] = (bf16_t)(rv * __expf(p));  // exp <= 1
        const float incl = p + wl;
        const float kh = kv * __expf(-incl);    // exp <= e^85, in range
        kh16[t][c] = (bf16_t)kh;
        khf[t] = kh;
        vtr[c][t] = (bf16_t)vv;
        p = incl;
      }
      const float wdv = __expf(p);  // exp(sum of w) <= 1
      wd[c] = wdv;
#pragma unroll
      for (int t = 0; t < 16; ++t) ktr[c][t] = (bf16_t)(khf[t] * wdv);
    } else {
      // threads 64..127: partial sums of diag[t] = sum_k r*u*k
      const int j = tid - 64, tt = j & 15, part = j >> 4, kp = part * 16;
      const size_t base = ((size_t)bh * Tc + t0 + tt) * Kc + kp;
      float s = 0.f;
#pragma unroll
      for (int q = 0; q < 16; ++q) s += r[base + q] * ulds[kp + q] * k[base + q];
      dpart[part][tt] = s;
    }
    __syncthreads();
    if (tid < 16)
      diag[tid] = dpart[0][tid] + dpart[1][tid] + dpart[2][tid] + dpart[3][tid];
    __syncthreads();

    // ---- phase B: per-wave WMMA work on a 16-wide V slice ----
    // A = r~ @ k^T  (16x16, reduce over k=64)
    v8f aacc = {0.f, 0.f, 0.f, 0.f, 0.f, 0.f, 0.f, 0.f};
#pragma unroll
    for (int kk = 0; kk < 64; kk += 32) {
      v16bf aop = load_op(&rt16[m][0], kk, half);
      v16bf bop = load_op(&kh16[m][0], kk, half);
      aacc = wmma_bf16(aop, bop, aacc);
    }
    // tril(A,-1) + diag, store to per-wave LDS buffer for reuse as A-operand
#pragma unroll
    for (int d = 0; d < 8; ++d) {
      const int t = d + 8 * half;
      float val = aacc[d];
      val = (t > m) ? val : ((t == m) ? diag[t] : 0.f);
      Am[wave][t][m] = (bf16_t)val;
    }

    // out = Am @ V  +  r~ @ S   (state BEFORE update)
    v8f oacc = {0.f, 0.f, 0.f, 0.f, 0.f, 0.f, 0.f, 0.f};
    {
      v16bf aop = load_op(&Am[wave][m][0], 0, half);   // K = s (padded to 32)
      v16bf bop = load_op(&vtr[v0 + m][0], 0, half);
      oacc = wmma_bf16(aop, bop, oacc);
    }
#pragma unroll
    for (int kk = 0; kk < 64; kk += 32) {
      v16bf aop = load_op(&rt16[m][0], kk, half);
      v16bf bop = load_op(&Sb[wave][m][0], kk, half);  // Sb is [v][k]
      oacc = wmma_bf16(aop, bop, oacc);
    }
#pragma unroll
    for (int d = 0; d < 8; ++d) {
      const int t = d + 8 * half;
      out[((size_t)bh * Tc + t0 + t) * Vc + v0 + m] = oacc[d];
    }

    // state update: S[v][k] = S*exp(sum w)[k] + (V^T @ K~)[v][k]
    const v16bf vop = load_op(&vtr[v0 + m][0], 0, half);  // A-op: V^T, K = s
#pragma unroll
    for (int kb = 0; kb < 64; kb += 16) {
      v16bf bop = load_op(&ktr[kb + m][0], 0, half);      // B-op: K~, [k][s]
      v8f dacc = {0.f, 0.f, 0.f, 0.f, 0.f, 0.f, 0.f, 0.f};
      dacc = wmma_bf16(vop, bop, dacc);
      const float wdk = wd[kb + m];
#pragma unroll
      for (int d = 0; d < 8; ++d) {
        const int vl = d + 8 * half;
        const float ns = St[wave][vl][kb + m] * wdk + dacc[d];
        St[wave][vl][kb + m] = ns;
        Sb[wave][vl][kb + m] = (bf16_t)ns;
      }
    }
    __syncthreads();
  }

  // ---- final state (B,H,K,V) ----
  for (int q = lane; q < 16 * 64; q += 32) {
    const int kk = q & 63, vl = q >> 6;
    s_out[(size_t)bh * Kc * Vc + (size_t)kk * Vc + v0 + vl] = St[wave][vl][kk];
  }
}

extern "C" void kernel_launch(void* const* d_in, const int* in_sizes, int n_in,
                              void* d_out, int out_size, void* d_ws,
                              size_t ws_size, hipStream_t stream) {
  (void)in_sizes; (void)n_in; (void)out_size; (void)d_ws; (void)ws_size;
  const float* r = (const float*)d_in[0];
  const float* k = (const float*)d_in[1];
  const float* v = (const float*)d_in[2];
  const float* w = (const float*)d_in[3];
  const float* u = (const float*)d_in[4];
  const float* s = (const float*)d_in[5];
  float* out = (float*)d_out;
  float* s_out = out + OUT_ELEMS;
  hipLaunchKernelGGL(rwkv6_wkv_kernel, dim3(Bc * Hc), dim3(128), 0, stream,
                     r, k, v, w, u, s, out, s_out);
}
